// NeuronSolarOpenDecoderLayer_62843961475295
// MI455X (gfx1250) — matrix operand order repaired and checked by reading, not compiled, not run on hardware
//
#include <hip/hip_runtime.h>
#include <hip/hip_bf16.h>

// ---------------------------------------------------------------------------
// CDNA5 (gfx1250) fused decoder layer: RMSNorm -> QKV (WMMA bf16) -> YaRN RoPE
// -> causal GQA flash-attention (WMMA) -> WO+residual -> RMSNorm -> sigmoid
// top-2 router -> gathered MoE (fused gate/up WMMA, silu, down WMMA w/ scaled
// atomic scatter) -> shared expert -> final residual add.
// Weight tiles are fp32 in HBM, read exactly once, staged to LDS as bf16.
// GLOBAL_LOAD_ASYNC_TO_LDS_B128 (ASYNCcnt) double-buffers the raw weight tile
// so the DMA overlaps conversion + v_wmma compute.
// ---------------------------------------------------------------------------

#if defined(__HIP_DEVICE_COMPILE__) && \
    __has_builtin(__builtin_amdgcn_global_load_async_to_lds_b128)
#define USE_ASYNC_LDS 1
#else
#define USE_ASYNC_LDS 0
#endif

typedef __attribute__((ext_vector_type(16))) __bf16 v16bf;
typedef __attribute__((ext_vector_type(8)))  float  v8f;
typedef int vint4 __attribute__((vector_size(16)));   // builtin's b128 unit

union BF16Frag { v16bf v; uint4 q[2]; unsigned short s[16]; };

__device__ __forceinline__ unsigned short f2bf(float f) {
  return __builtin_bit_cast(unsigned short, (__bf16)f);  // fptrunc, RNE
}
__device__ __forceinline__ float bf2f(unsigned short s) {
  return __uint_as_float(((unsigned)s) << 16);
}
__device__ __forceinline__ v8f wmma_bf16(v16bf a, v16bf b, v8f c) {
  return __builtin_amdgcn_wmma_f32_16x16x32_bf16(false, a, false, b, (short)0, c,
                                                 false, false);
}

// A fragment (16x32 bf16, row-major source, contiguous 2x16B per lane):
// lane<16: M=lane, K 0..15 sequential; lane>=16: M=lane-16, K 16..31
__device__ __forceinline__ v16bf load_afrag(const unsigned short* row, int kbase,
                                            int lane) {
  int c0 = kbase + ((lane >> 4) << 4);
  BF16Frag f;
  f.q[0] = *(const uint4*)(row + c0);
  f.q[1] = *(const uint4*)(row + c0 + 8);
  return f.v;
}
// B fragment (32x16 bf16, column-contiguous source):
// lane<16: N=lane, K 0..15 sequential; lane>=16: N=lane-16, K 16..31
__device__ __forceinline__ v16bf load_bfrag(const unsigned short* col, int kbase,
                                            int lane) {
  int c0 = kbase + ((lane >> 4) << 4);
  BF16Frag f;
  f.q[0] = *(const uint4*)(col + c0);
  f.q[1] = *(const uint4*)(col + c0 + 8);
  return f.v;
}

// ---- B tile staging: 32(K) x 128(N) fp32 -> LDS bf16 fragment order --------
__device__ __forceinline__ void frag_store4(unsigned short* frag, int k, int n4,
                                            float4 v) {
  int lanebase = (n4 & 15) + ((k >> 4) << 4);
  int rbase = ((n4 >> 4) << 9) + (k & 15);
  frag[rbase + ((lanebase + 0) << 4)] = f2bf(v.x);
  frag[rbase + ((lanebase + 1) << 4)] = f2bf(v.y);
  frag[rbase + ((lanebase + 2) << 4)] = f2bf(v.z);
  frag[rbase + ((lanebase + 3) << 4)] = f2bf(v.w);
}
// Direct path: global float4 loads -> convert -> frag LDS.
__device__ __forceinline__ void stage_direct(const float* __restrict__ B,
                                             int ldb, unsigned short* frag,
                                             int tid) {
#pragma unroll
  for (int c = 0; c < 4; ++c) {
    int chunk = tid + (c << 8);              // 0..1023 16B chunks
    int k = chunk >> 5, n4 = (chunk & 31) << 2;
    float4 v = *(const float4*)(B + (size_t)k * ldb + n4);
    frag_store4(frag, k, n4, v);
  }
}

#if USE_ASYNC_LDS
// Async DMA of raw fp32 tile into LDS (row-major [32][128]); ASYNCcnt-tracked.
__device__ __forceinline__ void async_copy_tile(const float* __restrict__ B,
                                                int ldb, float* raw, int tid) {
#pragma unroll
  for (int c = 0; c < 4; ++c) {
    int chunk = tid + (c << 8);
    int k = chunk >> 5, n4 = (chunk & 31) << 2;
    const float* ga = B + (size_t)k * ldb + n4;
    float* la = raw + k * 128 + n4;
    __builtin_amdgcn_global_load_async_to_lds_b128(
        (__attribute__((address_space(1))) vint4*)(void*)ga,
        (__attribute__((address_space(3))) vint4*)(void*)la, 0, 0);
  }
}
__device__ __forceinline__ void wait_async0() {
#if __has_builtin(__builtin_amdgcn_s_wait_asynccnt)
  __builtin_amdgcn_s_wait_asynccnt(0);
#else
  asm volatile("s_wait_asynccnt 0x0" ::: "memory");
#endif
}
// Convert raw LDS fp32 tile -> bf16 fragment LDS.
__device__ __forceinline__ void convert_raw_to_frag(const float* raw,
                                                    unsigned short* frag,
                                                    int tid) {
#pragma unroll
  for (int c = 0; c < 4; ++c) {
    int chunk = tid + (c << 8);
    int k = chunk >> 5, n4 = (chunk & 31) << 2;
    float4 v = *(const float4*)(raw + k * 128 + n4);
    frag_store4(frag, k, n4, v);
  }
}
#endif

// ---------------------------------------------------------------------------
// Generic GEMM: C[M,N] = A_bf16[M,K] @ B_f32[K,N] (+ addsrc). M,N mult of 128.
// ---------------------------------------------------------------------------
__global__ __launch_bounds__(256) void gemm_plain(
    const unsigned short* __restrict__ A, const float* __restrict__ B,
    float* __restrict__ C, const float* __restrict__ addsrc, int N, int K) {
  __shared__ __align__(32) unsigned short fragB[4096];
#if USE_ASYNC_LDS
  __shared__ __align__(16) float rawB[2][32 * 128];
#endif
  const int tid = threadIdx.x, lane = tid & 31, wave = tid >> 5;
  const int nbase = blockIdx.x * 128, mbase = blockIdx.y * 128;
  const unsigned short* Arow = A + (size_t)(mbase + wave * 16 + (lane & 15)) * K;
  const v8f vz = {0.f, 0.f, 0.f, 0.f, 0.f, 0.f, 0.f, 0.f};
  v8f acc[8];
#pragma unroll
  for (int i = 0; i < 8; ++i) acc[i] = vz;
#if USE_ASYNC_LDS
  async_copy_tile(B + nbase, N, rawB[0], tid);
  int cur = 0;
#endif
  for (int kb = 0; kb < K; kb += 32) {
#if USE_ASYNC_LDS
    wait_async0();
    __syncthreads();
    if (kb + 32 < K)
      async_copy_tile(B + (size_t)(kb + 32) * N + nbase, N, rawB[cur ^ 1], tid);
    convert_raw_to_frag(rawB[cur], fragB, tid);
    cur ^= 1;
#else
    __syncthreads();
    stage_direct(B + (size_t)kb * N + nbase, N, fragB, tid);
    if (kb + 32 < K)
      __builtin_prefetch(B + (size_t)(kb + 32) * N + nbase + tid, 0, 0);
#endif
    __syncthreads();
    v16bf a = load_afrag(Arow, kb, lane);
#pragma unroll
    for (int nx = 0; nx < 8; ++nx) {
      v16bf b = *(const v16bf*)&fragB[(nx << 9) + (lane << 4)];
      acc[nx] = wmma_bf16(a, b, acc[nx]);
    }
  }
  const int r0 = mbase + wave * 16 + ((lane >> 4) << 3);
  const int c0 = nbase + (lane & 15);
#pragma unroll
  for (int nx = 0; nx < 8; ++nx)
#pragma unroll
    for (int r = 0; r < 8; ++r) {
      size_t o = (size_t)(r0 + r) * N + c0 + nx * 16;
      float v = acc[nx][r];
      if (addsrc) v += addsrc[o];
      C[o] = v;
    }
}

// ---------------------------------------------------------------------------
// Fused gate/up GEMM (gathered rows): Act = silu(A@Bg) * (A@Bu), bf16 out.
// ---------------------------------------------------------------------------
__global__ __launch_bounds__(256) void gemm_gateup(
    const unsigned short* __restrict__ A, const float* __restrict__ Bg0,
    const float* __restrict__ Bu0, size_t bstride,
    unsigned short* __restrict__ Act, const int* __restrict__ toklist,
    const int* __restrict__ offE, int Mfixed, int N, int K) {
  __shared__ __align__(32) unsigned short fragG[4096];
  __shared__ __align__(32) unsigned short fragU[4096];
#if USE_ASYNC_LDS
  __shared__ __align__(16) float rawG[2][32 * 128];
  __shared__ __align__(16) float rawU[2][32 * 128];
#endif
  const int e = blockIdx.z;
  const int rowoff = offE ? offE[e] : 0;
  const int cnt = offE ? (offE[e + 1] - offE[e]) : Mfixed;
  const int mbase = blockIdx.y * 128;
  if (mbase >= cnt) return;
  const float* Bg = Bg0 + (size_t)e * bstride;
  const float* Bu = Bu0 + (size_t)e * bstride;
  const int tid = threadIdx.x, lane = tid & 31, wave = tid >> 5;
  const int nbase = blockIdx.x * 128;
  const int s = mbase + wave * 16 + (lane & 15);
  const int tok = (s < cnt) ? (toklist ? toklist[rowoff + s] : s) : 0;
  const unsigned short* Arow = A + (size_t)tok * K;
  const v8f vz = {0.f, 0.f, 0.f, 0.f, 0.f, 0.f, 0.f, 0.f};
  v8f accg[8], accu[8];
#pragma unroll
  for (int i = 0; i < 8; ++i) { accg[i] = vz; accu[i] = vz; }
#if USE_ASYNC_LDS
  async_copy_tile(Bg + nbase, N, rawG[0], tid);
  async_copy_tile(Bu + nbase, N, rawU[0], tid);
  int cur = 0;
#endif
  for (int kb = 0; kb < K; kb += 32) {
#if USE_ASYNC_LDS
    wait_async0();
    __syncthreads();
    if (kb + 32 < K) {
      async_copy_tile(Bg + (size_t)(kb + 32) * N + nbase, N, rawG[cur ^ 1], tid);
      async_copy_tile(Bu + (size_t)(kb + 32) * N + nbase, N, rawU[cur ^ 1], tid);
    }
    convert_raw_to_frag(rawG[cur], fragG, tid);
    convert_raw_to_frag(rawU[cur], fragU, tid);
    cur ^= 1;
#else
    __syncthreads();
    stage_direct(Bg + (size_t)kb * N + nbase, N, fragG, tid);
    stage_direct(Bu + (size_t)kb * N + nbase, N, fragU, tid);
    if (kb + 32 < K)
      __builtin_prefetch(Bg + (size_t)(kb + 32) * N + nbase + tid, 0, 0);
#endif
    __syncthreads();
    v16bf a = load_afrag(Arow, kb, lane);
#pragma unroll
    for (int nx = 0; nx < 8; ++nx) {
      v16bf bg = *(const v16bf*)&fragG[(nx << 9) + (lane << 4)];
      v16bf bu = *(const v16bf*)&fragU[(nx << 9) + (lane << 4)];
      accg[nx] = wmma_bf16(a, bg, accg[nx]);
      accu[nx] = wmma_bf16(a, bu, accu[nx]);
    }
  }
  const int rbase = mbase + wave * 16 + ((lane >> 4) << 3);
#pragma unroll
  for (int r = 0; r < 8; ++r) {
    int srow = rbase + r;
    if (srow < cnt) {
#pragma unroll
      for (int nx = 0; nx < 8; ++nx) {
        float g = accg[nx][r], u = accu[nx][r];
        float val = g * (1.f / (1.f + __expf(-g))) * u;  // silu(g)*u
        Act[(size_t)(rowoff + srow) * N + nbase + nx * 16 + (lane & 15)] =
            f2bf(val);
      }
    }
  }
}

// ---------------------------------------------------------------------------
// Down GEMM: scatter-add (optionally weighted+atomic) into accum[T,N].
// ---------------------------------------------------------------------------
__global__ __launch_bounds__(256) void gemm_down(
    const unsigned short* __restrict__ Act, const float* __restrict__ B0,
    size_t bstride, float* __restrict__ Out, const int* __restrict__ toklist,
    const float* __restrict__ wslot, const int* __restrict__ offE, int Mfixed,
    int N, int K, int use_atomic) {
  __shared__ __align__(32) unsigned short fragB[4096];
#if USE_ASYNC_LDS
  __shared__ __align__(16) float rawB[2][32 * 128];
#endif
  const int e = blockIdx.z;
  const int rowoff = offE ? offE[e] : 0;
  const int cnt = offE ? (offE[e + 1] - offE[e]) : Mfixed;
  const int mbase = blockIdx.y * 128;
  if (mbase >= cnt) return;
  const float* B = B0 + (size_t)e * bstride;
  const int tid = threadIdx.x, lane = tid & 31, wave = tid >> 5;
  const int nbase = blockIdx.x * 128;
  const int s = mbase + wave * 16 + (lane & 15);
  const unsigned short* Arow = Act + (size_t)(rowoff + ((s < cnt) ? s : 0)) * K;
  const v8f vz = {0.f, 0.f, 0.f, 0.f, 0.f, 0.f, 0.f, 0.f};
  v8f acc[8];
#pragma unroll
  for (int i = 0; i < 8; ++i) acc[i] = vz;
#if USE_ASYNC_LDS
  async_copy_tile(B + nbase, N, rawB[0], tid);
  int cur = 0;
#endif
  for (int kb = 0; kb < K; kb += 32) {
#if USE_ASYNC_LDS
    wait_async0();
    __syncthreads();
    if (kb + 32 < K)
      async_copy_tile(B + (size_t)(kb + 32) * N + nbase, N, rawB[cur ^ 1], tid);
    convert_raw_to_frag(rawB[cur], fragB, tid);
    cur ^= 1;
#else
    __syncthreads();
    stage_direct(B + (size_t)kb * N + nbase, N, fragB, tid);
    if (kb + 32 < K)
      __builtin_prefetch(B + (size_t)(kb + 32) * N + nbase + tid, 0, 0);
#endif
    __syncthreads();
    v16bf a = load_afrag(Arow, kb, lane);
#pragma unroll
    for (int nx = 0; nx < 8; ++nx) {
      v16bf b = *(const v16bf*)&fragB[(nx << 9) + (lane << 4)];
      acc[nx] = wmma_bf16(a, b, acc[nx]);
    }
  }
  const int rbase = mbase + wave * 16 + ((lane >> 4) << 3);
#pragma unroll
  for (int r = 0; r < 8; ++r) {
    int srow = rbase + r;
    if (srow < cnt) {
      int tok; float wgt;
      if (toklist) { tok = toklist[rowoff + srow]; wgt = wslot[rowoff + srow]; }
      else         { tok = srow; wgt = 1.f; }
#pragma unroll
      for (int nx = 0; nx < 8; ++nx) {
        float v = acc[nx][r] * wgt;
        size_t o = (size_t)tok * N + nbase + nx * 16 + (lane & 15);
        if (use_atomic) unsafeAtomicAdd(&Out[o], v);  // global_atomic_add_f32
        else            Out[o] += v;
      }
    }
  }
}

// ---------------------------------------------------------------------------
// Causal GQA flash attention. One block = 1 head x 128 queries (8 waves x 16).
// ---------------------------------------------------------------------------
__global__ __launch_bounds__(256) void attn_kernel(
    const unsigned short* __restrict__ Qbf,   // [S, 32*128]
    const unsigned short* __restrict__ Kbf,   // [8, S, 128]
    const unsigned short* __restrict__ VT,    // [8, 128, S]
    unsigned short* __restrict__ CtxBf) {     // [S, 32*128]
  __shared__ __align__(32) unsigned short ldsP[8 * 512];
  const int lane = threadIdx.x & 31, wave = threadIdx.x >> 5;
  const int h = blockIdx.y, kvh = h >> 2;
  const int qbase = blockIdx.x * 128 + wave * 16;
  const float scale = 0.08838834764831845f;   // 1/sqrt(128)

  const unsigned short* Qrow =
      Qbf + (size_t)(qbase + (lane & 15)) * 4096 + h * 128;
  v16bf qf[4];
#pragma unroll
  for (int kk = 0; kk < 4; ++kk) qf[kk] = load_afrag(Qrow, kk * 32, lane);

  const v8f vz = {0.f, 0.f, 0.f, 0.f, 0.f, 0.f, 0.f, 0.f};
  v8f o[8];
#pragma unroll
  for (int i = 0; i < 8; ++i) o[i] = vz;
  float mr[8], lr[8];
#pragma unroll
  for (int r = 0; r < 8; ++r) { mr[r] = -1e30f; lr[r] = 0.f; }

  const unsigned short* Kb = Kbf + (size_t)kvh * 1024 * 128;
  const unsigned short* Vb = VT + (size_t)kvh * 128 * 1024;
  unsigned short* myP = ldsP + wave * 512;
  const int nt = (qbase + 47) >> 5;           // causal #32-key tiles
  const int row0 = qbase + ((lane >> 4) << 3);

  for (int kt = 0; kt < nt; ++kt) {
    const int jb = kt << 5;
    v8f sf0 = vz, sf1 = vz;
    {
      const unsigned short* K0 = Kb + (size_t)(jb + (lane & 15)) * 128;
      const unsigned short* K1 = K0 + 16 * 128;
#pragma unroll
      for (int kk = 0; kk < 4; ++kk) {
        sf0 = wmma_bf16(qf[kk], load_bfrag(K0, kk * 32, lane), sf0);
        sf1 = wmma_bf16(qf[kk], load_bfrag(K1, kk * 32, lane), sf1);
      }
    }
    const int col0 = jb + (lane & 15);
    float p0[8], p1[8];
#pragma unroll
    for (int r = 0; r < 8; ++r) {
      int row = row0 + r;
      p0[r] = (col0 <= row) ? sf0[r] * scale : -1e30f;
      p1[r] = (col0 + 16 <= row) ? sf1[r] * scale : -1e30f;
    }
#pragma unroll
    for (int r = 0; r < 8; ++r) {
      float rm = fmaxf(p0[r], p1[r]);
#pragma unroll
      for (int d = 1; d < 16; d <<= 1) rm = fmaxf(rm, __shfl_xor(rm, d, 32));
      float mnew = fmaxf(mr[r], rm);
      float alpha = __expf(mr[r] - mnew);
      mr[r] = mnew;
      float e0 = __expf(p0[r] - mnew);
      float e1 = __expf(p1[r] - mnew);
      int m = r + ((lane >> 4) << 3);
      myP[m * 32 + (lane & 15)] = f2bf(e0);
      myP[m * 32 + 16 + (lane & 15)] = f2bf(e1);
      float rs = e0 + e1;
#pragma unroll
      for (int d = 1; d < 16; d <<= 1) rs += __shfl_xor(rs, d, 32);
      lr[r] = lr[r] * alpha + rs;
#pragma unroll
      for (int nx = 0; nx < 8; ++nx) o[nx][r] *= alpha;
    }
    // Re-read P as A-fragment (per-wave LDS region; DScnt orders RAW in-wave).
    const unsigned short* pr = myP + (lane & 15) * 32;
    v16bf pf = load_afrag(pr, 0, lane);
#pragma unroll
    for (int nx = 0; nx < 8; ++nx) {
      const unsigned short* Vrow =
          Vb + (size_t)(nx * 16 + (lane & 15)) * 1024 + jb;
      o[nx] = wmma_bf16(pf, load_bfrag(Vrow, 0, lane), o[nx]);
    }
  }
#pragma unroll
  for (int r = 0; r < 8; ++r) {
    float inv = 1.f / lr[r];
    int row = row0 + r;
#pragma unroll
    for (int nx = 0; nx < 8; ++nx)
      CtxBf[(size_t)row * 4096 + h * 128 + nx * 16 + (lane & 15)] =
          f2bf(o[nx][r] * inv);
  }
}

// ---------------------------------------------------------------------------
// Small helper kernels
// ---------------------------------------------------------------------------
__global__ __launch_bounds__(256) void rmsnorm_kernel(
    const float* __restrict__ X, const float* __restrict__ W,
    unsigned short* __restrict__ Obf, int Hd) {
  __shared__ float red[256];
  const int t = blockIdx.x;
  const float* xr = X + (size_t)t * Hd;
  float ss = 0.f;
  for (int j = threadIdx.x; j < Hd; j += 256) { float v = xr[j]; ss += v * v; }
  red[threadIdx.x] = ss;
  __syncthreads();
  for (int s = 128; s > 0; s >>= 1) {
    if (threadIdx.x < s) red[threadIdx.x] += red[threadIdx.x + s];
    __syncthreads();
  }
  float rs = rsqrtf(red[0] / (float)Hd + 1e-5f);
  for (int j = threadIdx.x; j < Hd; j += 256)
    Obf[(size_t)t * Hd + j] = f2bf(xr[j] * rs * W[j]);
}

__device__ __forceinline__ void yarn_cs(float p, int i, float* c, float* s) {
  const float lnb = 13.815510558f;                       // ln(1e6)
  float ib = __expf(-(float)i * (lnb / 64.f));           // 1e6^(-i/64)
  float low = floorf(64.f * __logf(65536.f / (32.f * 6.28318530718f)) / lnb);
  float high = ceilf(64.f * __logf(65536.f / 6.28318530718f) / lnb);
  low = fmaxf(low, 0.f); high = fminf(high, 127.f);
  float ramp = ((float)i - low) / (high - low);
  float extra = 1.f - fminf(fmaxf(ramp, 0.f), 1.f);
  float inv = (ib * 0.5f) * (1.f - extra) + ib * extra;  // factor = 2
  float ang = p * inv, sv, cv;
  __sincosf(ang, &sv, &cv);
  const float scal = 1.0693147180559945f;                // 0.1*ln(2)+1
  *c = cv * scal; *s = sv * scal;
}

__global__ __launch_bounds__(256) void rope_q_kernel(
    const float* __restrict__ Q, unsigned short* __restrict__ Qbf) {
  int gid = blockIdx.x * 256 + threadIdx.x;   // S*H*64 threads
  int t = gid >> 11, rem = gid & 2047;
  int hh = rem >> 6, i = rem & 63;
  size_t base = (size_t)t * 4096 + hh * 128 + i;
  float q1 = Q[base], q2 = Q[base + 64], c, s;
  yarn_cs((float)t, i, &c, &s);
  Qbf[base] = f2bf(q1 * c - q2 * s);
  Qbf[base + 64] = f2bf(q2 * c + q1 * s);
}

__global__ __launch_bounds__(256) void rope_k_kernel(
    const float* __restrict__ K, unsigned short* __restrict__ Kbf) {
  int gid = blockIdx.x * 256 + threadIdx.x;   // S*HKV*64 threads
  int t = gid >> 9, rem = gid & 511;
  int kvh = rem >> 6, i = rem & 63;
  size_t src = (size_t)t * 1024 + kvh * 128 + i;
  size_t dst = ((size_t)kvh * 1024 + t) * 128 + i;
  float k1 = K[src], k2 = K[src + 64], c, s;
  yarn_cs((float)t, i, &c, &s);
  Kbf[dst] = f2bf(k1 * c - k2 * s);
  Kbf[dst + 64] = f2bf(k2 * c + k1 * s);
}

__global__ __launch_bounds__(256) void transpose_v_kernel(
    const float* __restrict__ V, unsigned short* __restrict__ VT) {
  int gid = blockIdx.x * 256 + threadIdx.x;   // S*HKV*128 threads
  int t = gid >> 10, r = gid & 1023;
  int kvh = r >> 7, d = r & 127;
  VT[((size_t)kvh * 128 + d) * 1024 + t] = f2bf(V[(size_t)t * 1024 + r]);
}

__global__ __launch_bounds__(256) void router_kernel(
    const unsigned short* __restrict__ Hbf, const float* __restrict__ Wr,
    const float* __restrict__ ebias, int* __restrict__ topidx,
    float* __restrict__ topw, int* __restrict__ cnt, int Hd) {
  __shared__ float red[256 * 8];
  const int t = blockIdx.x;
  float acc[8] = {0.f, 0.f, 0.f, 0.f, 0.f, 0.f, 0.f, 0.f};
  for (int j = threadIdx.x; j < Hd; j += 256) {
    float hv = bf2f(Hbf[(size_t)t * Hd + j]);
    const float* wr = &Wr[(size_t)j * 8];
#pragma unroll
    for (int e = 0; e < 8; ++e) acc[e] += hv * wr[e];
  }
#pragma unroll
  for (int e = 0; e < 8; ++e) red[threadIdx.x * 8 + e] = acc[e];
  __syncthreads();
  for (int s = 128; s > 0; s >>= 1) {
    if (threadIdx.x < s)
#pragma unroll
      for (int e = 0; e < 8; ++e)
        red[threadIdx.x * 8 + e] += red[(threadIdx.x + s) * 8 + e];
    __syncthreads();
  }
  if (threadIdx.x == 0) {
    float sig[8], sel[8];
#pragma unroll
    for (int e = 0; e < 8; ++e) {
      sig[e] = 1.f / (1.f + __expf(-red[e]));
      sel[e] = sig[e] + ebias[e];
    }
    int i0 = 0;
    for (int e = 1; e < 8; ++e) if (sel[e] > sel[i0]) i0 = e;
    int i1 = -1;
    for (int e = 0; e < 8; ++e)
      if (e != i0 && (i1 < 0 || sel[e] > sel[i1])) i1 = e;
    float w0 = sig[i0], w1 = sig[i1], inv = 1.f / (w0 + w1);
    topidx[t * 2] = i0; topidx[t * 2 + 1] = i1;
    topw[t * 2] = w0 * inv; topw[t * 2 + 1] = w1 * inv;
    atomicAdd(&cnt[i0], 1); atomicAdd(&cnt[i1], 1);
  }
}

// Deterministic prefix offsets + gathered token lists (single thread).
__global__ void route_build(const int* __restrict__ cnt,
                            const int* __restrict__ topidx,
                            const float* __restrict__ topw,
                            int* __restrict__ offE, int* __restrict__ toklist,
                            float* __restrict__ wslot, int T) {
  int run[9]; run[0] = 0;
  for (int e = 0; e < 8; ++e) run[e + 1] = run[e] + cnt[e];
  for (int e = 0; e <= 8; ++e) offE[e] = run[e];
  int cur[8];
  for (int e = 0; e < 8; ++e) cur[e] = run[e];
  for (int t = 0; t < T; ++t)
    for (int j = 0; j < 2; ++j) {
      int e = topidx[t * 2 + j];
      int r = cur[e]++;
      toklist[r] = t;
      wslot[r] = topw[t * 2 + j];
    }
}

__global__ __launch_bounds__(256) void final_add_kernel(
    const float* __restrict__ x2, const float* __restrict__ accum,
    float* __restrict__ out) {
  int gid = blockIdx.x * 256 + threadIdx.x;
  out[gid] = x2[gid] + accum[gid];
}

// ---------------------------------------------------------------------------
extern "C" void kernel_launch(void* const* d_in, const int* in_sizes, int n_in,
                              void* d_out, int out_size, void* d_ws,
                              size_t ws_size, hipStream_t stream) {
  (void)in_sizes; (void)n_in; (void)out_size; (void)ws_size;
  const float* x        = (const float*)d_in[0];
  // d_in[1] = position_ids == arange(S): positions derived as t in-kernel.
  const float* w_norm1  = (const float*)d_in[2];
  const float* wq       = (const float*)d_in[3];
  const float* wk       = (const float*)d_in[4];
  const float* wv       = (const float*)d_in[5];
  const float* wo       = (const float*)d_in[6];
  const float* w_norm2  = (const float*)d_in[7];
  const float* w_router = (const float*)d_in[8];
  const float* e_bias   = (const float*)d_in[9];
  const float* w_gate   = (const float*)d_in[10];
  const float* w_up     = (const float*)d_in[11];
  const float* w_down   = (const float*)d_in[12];
  const float* ws_gate  = (const float*)d_in[13];
  const float* ws_up    = (const float*)d_in[14];
  const float* ws_down  = (const float*)d_in[15];
  float* out = (float*)d_out;

  constexpr int S = 1024, Hd = 4096, H = 32, HKV = 8, E = 8, IM = 1280;
  constexpr size_t MB = 1024 * 1024;
  char* base = (char*)d_ws;
  // Region map (with lifetime-based reuse):
  unsigned short* h_bf   = (unsigned short*)(base + 0 * MB);   // 8MB
  unsigned short* ctx_bf = h_bf;                               // reuse
  float*          qbuf   = (float*)(base + 8 * MB);            // 16MB
  float*          x2     = qbuf;                               // reuse
  float*          kbuf   = (float*)(base + 24 * MB);           // 4MB
  float*          vbuf   = (float*)(base + 28 * MB);           // 4MB
  unsigned short* h2_bf  = (unsigned short*)(base + 24 * MB);  // reuse k/v
  unsigned short* q_bf   = (unsigned short*)(base + 32 * MB);  // 8MB
  unsigned short* act_bf = q_bf;                               // reuse
  unsigned short* k_bf   = (unsigned short*)(base + 40 * MB);  // 2MB
  unsigned short* vT_bf  = (unsigned short*)(base + 42 * MB);  // 2MB
  unsigned short* sact_bf = k_bf;                              // reuse (2.5MB)
  float*          accum  = (float*)(base + 44 * MB);           // 16MB
  char* small = base + 60 * MB;
  int*   topidx  = (int*)(small);
  float* topw    = (float*)(small + 8192);
  int*   cnt     = (int*)(small + 16384);
  int*   offE    = (int*)(small + 16384 + 64);
  int*   toklist = (int*)(small + 16384 + 128);
  float* wslot   = (float*)(small + 16384 + 128 + 8192);

  (void)hipMemsetAsync(accum, 0, (size_t)S * Hd * sizeof(float), stream);
  (void)hipMemsetAsync(cnt, 0, 64, stream);

  // ---- attention block ----
  rmsnorm_kernel<<<S, 256, 0, stream>>>(x, w_norm1, h_bf, Hd);
  gemm_plain<<<dim3(Hd / 128, S / 128), 256, 0, stream>>>(h_bf, wq, qbuf,
                                                          nullptr, Hd, Hd);
  gemm_plain<<<dim3(1024 / 128, S / 128), 256, 0, stream>>>(h_bf, wk, kbuf,
                                                            nullptr, 1024, Hd);
  gemm_plain<<<dim3(1024 / 128, S / 128), 256, 0, stream>>>(h_bf, wv, vbuf,
                                                            nullptr, 1024, Hd);
  rope_q_kernel<<<(S * H * 64) / 256, 256, 0, stream>>>(qbuf, q_bf);
  rope_k_kernel<<<(S * HKV * 64) / 256, 256, 0, stream>>>(kbuf, k_bf);
  transpose_v_kernel<<<(S * HKV * 128) / 256, 256, 0, stream>>>(vbuf, vT_bf);
  attn_kernel<<<dim3(S / 128, H), 256, 0, stream>>>(q_bf, k_bf, vT_bf, ctx_bf);
  gemm_plain<<<dim3(Hd / 128, S / 128), 256, 0, stream>>>(ctx_bf, wo, x2, x,
                                                          Hd, Hd);

  // ---- MoE block ----
  rmsnorm_kernel<<<S, 256, 0, stream>>>(x2, w_norm2, h2_bf, Hd);
  router_kernel<<<S, 256, 0, stream>>>(h2_bf, w_router, e_bias, topidx, topw,
                                       cnt, Hd);
  route_build<<<1, 1, 0, stream>>>(cnt, topidx, topw, offE, toklist, wslot, S);
  gemm_gateup<<<dim3(IM / 128, S / 128, E), 256, 0, stream>>>(
      h2_bf, w_gate, w_up, (size_t)Hd * IM, act_bf, toklist, offE, 0, IM, Hd);
  gemm_down<<<dim3(Hd / 128, S / 128, E), 256, 0, stream>>>(
      act_bf, w_down, (size_t)IM * Hd, accum, toklist, wslot, offE, 0, Hd, IM,
      1);
  gemm_gateup<<<dim3(IM / 128, S / 128, 1), 256, 0, stream>>>(
      h2_bf, ws_gate, ws_up, 0, sact_bf, nullptr, nullptr, S, IM, Hd);
  gemm_down<<<dim3(Hd / 128, S / 128, 1), 256, 0, stream>>>(
      sact_bf, ws_down, 0, accum, nullptr, nullptr, nullptr, S, Hd, IM, 0);
  final_add_kernel<<<(S * Hd) / 256, 256, 0, stream>>>(x2, accum, out);
}
